// Net_88261577933457
// MI455X (gfx1250) — matrix operand order repaired and checked by reading, not compiled, not run on hardware
//
#include <hip/hip_runtime.h>
#include <hip/hip_bf16.h>

// ---------------- problem constants ----------------
#define BATCH 64
#define NRAW  511
#define NB    512            // padded node count (CLS + 511)
#define NF    32
#define DM    192            // model dim (= per-head dim)
#define NH    4
#define HD    768            // H * D = hidden of gat projection and FFN
#define ROWS  (BATCH * NB)   // 32768

typedef __attribute__((ext_vector_type(16))) __bf16 v16bf;
typedef __attribute__((ext_vector_type(8)))  __bf16 v8bf;
typedef __attribute__((ext_vector_type(8)))  float  v8f;

__device__ __forceinline__ __bf16 f2bf(float f) {
  unsigned u = __builtin_bit_cast(unsigned, f);
  unsigned r = (u + 0x7FFFu + ((u >> 16) & 1u)) >> 16;
  unsigned short s = (unsigned short)r;
  return __builtin_bit_cast(__bf16, s);
}

__device__ __forceinline__ v16bf mk16(v8bf lo, v8bf hi) {
  v16bf r;
#pragma unroll
  for (int i = 0; i < 8; ++i) { r[i] = lo[i]; r[i + 8] = hi[i]; }
  return r;
}

// ---------------- adjacency bitmask ----------------
__global__ __launch_bounds__(32) void mask_kernel(const float* __restrict__ adj,
                                                  unsigned* __restrict__ MASK) {
  int bi = blockIdx.x;            // b*NB + i
  int b = bi >> 9, i = bi & 511;
  int w = threadIdx.x;
  if (w >= 16) return;
  unsigned bits = 0u;
  for (int jj = 0; jj < 32; ++jj) {
    int j = w * 32 + jj;
    bool m;
    if (i == j)       m = true;
    else if (i == 0)  m = (j < NRAW);
    else if (j == 0)  m = false;
    else              m = adj[((size_t)b * NRAW + (i - 1)) * NRAW + (j - 1)] != 0.f;
    bits |= (m ? 1u : 0u) << jj;
  }
  MASK[(size_t)bi * 16 + w] = bits;
}

// ---------------- input embedding ----------------
__global__ __launch_bounds__(192) void embed_kernel(
    const float* __restrict__ x, const int* __restrict__ depth,
    const float* __restrict__ opW, const float* __restrict__ opb,
    const float* __restrict__ depW, const float* __restrict__ depb,
    const float* __restrict__ clsp, float* __restrict__ X) {
  int e = blockIdx.x;             // b*NRAW + n
  int b = e / NRAW, n = e % NRAW;
  int t = threadIdx.x;
  __shared__ float xs[NF];
  if (t < NF) xs[t] = x[(unsigned)((b * NRAW + n) * NF + t)];
  __syncthreads();
  int d = depth[(unsigned)(b * NRAW + n)];
  float acc = opb[t] + depb[t] + depW[(unsigned)(d * DM + t)];
#pragma unroll 8
  for (int k = 0; k < NF; ++k) acc += xs[k] * opW[(unsigned)(k * DM + t)];
  X[(size_t)(unsigned)((b * NB + (n + 1)) * DM + t)] = acc;
  if (n == 0) X[(size_t)(unsigned)((b * NB) * DM + t)] = clsp[t];   // CLS row
}

// ---------------- layer norm (row length 192) ----------------
__global__ __launch_bounds__(192) void ln_kernel(const float* __restrict__ in,
                                                 float* __restrict__ out,
                                                 const float* __restrict__ g,
                                                 const float* __restrict__ bb) {
  int row = blockIdx.x, t = threadIdx.x;
  __shared__ float rs[DM], rs2[DM];
  __shared__ float mean_s, rstd_s;
  float v = in[(size_t)(unsigned)(row * DM + t)];
  rs[t] = v; rs2[t] = v * v;
  __syncthreads();
  for (int off = 96; off >= 3; off >>= 1) {
    if (t < off) { rs[t] += rs[t + off]; rs2[t] += rs2[t + off]; }
    __syncthreads();
  }
  if (t == 0) {
    float m = (rs[0] + rs[1] + rs[2]) * (1.f / DM);
    float var = (rs2[0] + rs2[1] + rs2[2]) * (1.f / DM) - m * m;
    mean_s = m; rstd_s = rsqrtf(var + 1e-5f);
  }
  __syncthreads();
  out[(size_t)(unsigned)(row * DM + t)] = (v - mean_s) * rstd_s * g[t] + bb[t];
}

// ---------------- generic bf16 WMMA GEMM ----------------
// C[M,N] = A[M,K] @ B[K,N]  (+bias +res1 +res2, optional relu), all f32 in mem.
// Block tile: 64(M) x 64(N), K-step 32, 8 waves; each wave: 16x32 -> 2 accums,
// 2 WMMAs per K-step. Epilogue specialized at compile time (no branches).
#define TM 64
#define TN 64
#define TK 32
#define APITCH 40   // 32 + 8 pad, keeps 16B alignment of 8-elt chunks
template <int HAS_BIAS, int HAS_R1, int HAS_R2, int RELU>
__global__ __launch_bounds__(256) void gemm_bf16_wmma(
    const float* __restrict__ A, int lda,
    const float* __restrict__ Bm, int ldb,
    float* __restrict__ C, int ldc, int K,
    const float* __restrict__ bias,
    const float* __restrict__ res1,
    const float* __restrict__ res2) {
  __shared__ __bf16 As[TM][APITCH];
  __shared__ __bf16 Bs[TN][APITCH];   // stored transposed: Bs[n][k]
  int tid = threadIdx.x;
  int lane = tid & 31, wave = tid >> 5;
  int m0 = blockIdx.y * TM, n0 = blockIdx.x * TN;
  int wm = (wave & 3) * 16;           // wave M offset inside block tile
  int wn = (wave >> 2) * 32;          // wave N offset inside block tile
  int half = lane >> 4, rl = lane & 15;
  v8f acc0 = {0.f, 0.f, 0.f, 0.f, 0.f, 0.f, 0.f, 0.f};
  v8f acc1 = acc0;

  for (int k0 = 0; k0 < K; k0 += TK) {
    for (int e = tid; e < TM * TK; e += 256) {
      int r = e >> 5, c = e & 31;
      As[r][c] = f2bf(A[(size_t)(unsigned)((m0 + r) * lda + k0 + c)]);
    }
    for (int e = tid; e < TN * TK; e += 256) {
      int k = e >> 6, n = e & 63;
      Bs[n][k] = f2bf(Bm[(size_t)(unsigned)((k0 + k) * ldb + n0 + n)]);
    }
    if (k0 + TK < K) {  // prefetch next tiles into L2 (global_prefetch_b8)
      __builtin_prefetch(&A[(size_t)(unsigned)((m0 + (tid >> 2)) * lda + k0 + TK)], 0, 1);
      __builtin_prefetch(&Bm[(size_t)(unsigned)((k0 + TK + (tid >> 3)) * ldb + n0)], 0, 1);
    }
    __syncthreads();
    // A operand: row = wm+rl, 16 contiguous K at half*16
    v16bf a = mk16(*(const v8bf*)&As[wm + rl][half * 16],
                   *(const v8bf*)&As[wm + rl][half * 16 + 8]);
    // B operand: col, 16 contiguous K at half*16
    v16bf b0 = mk16(*(const v8bf*)&Bs[wn + rl][half * 16],
                    *(const v8bf*)&Bs[wn + rl][half * 16 + 8]);
    v16bf b1 = mk16(*(const v8bf*)&Bs[wn + 16 + rl][half * 16],
                    *(const v8bf*)&Bs[wn + 16 + rl][half * 16 + 8]);
    acc0 = __builtin_amdgcn_wmma_f32_16x16x32_bf16(false, a, false, b0,
                                                   (short)0, acc0, false, false);
    acc1 = __builtin_amdgcn_wmma_f32_16x16x32_bf16(false, a, false, b1,
                                                   (short)0, acc1, false, false);
    __syncthreads();
  }
  // C layout: VGPR v -> M = v + 8*half, N = lane%16
#pragma unroll
  for (int t = 0; t < 2; ++t) {
    int col = n0 + wn + t * 16 + rl;
    float bv = HAS_BIAS ? bias[col] : 0.f;
    v8f acc = t == 0 ? acc0 : acc1;
#pragma unroll
    for (int v = 0; v < 8; ++v) {
      int row = m0 + wm + v + 8 * half;
      unsigned idx = (unsigned)(row * ldc + col);
      float xv = acc[v] + bv;
      if (HAS_R1) xv += res1[(size_t)idx];
      if (HAS_R2) xv += res2[(size_t)idx];
      if (RELU)   xv = fmaxf(xv, 0.f);
      C[(size_t)idx] = xv;
    }
  }
}

// ---------------- attention prep ----------------
// Writes V in K-contiguous layout HBT[b][h][c][j] (j = NB contiguous) so the
// attention kernel can load WMMA B operands straight from global memory.
// Also computes a_src/a_dst head dots.
__global__ __launch_bounds__(192) void prep_attn(
    const float* __restrict__ Hf, const float* __restrict__ att_src,
    const float* __restrict__ att_dst, __bf16* __restrict__ HBT,
    float* __restrict__ ASRC, float* __restrict__ ADST) {
  int row = blockIdx.x;     // b*NB + j
  int b = row >> 9, j = row & 511;
  int t = threadIdx.x;
  __shared__ float red[DM];
  const float* hrow = Hf + (size_t)(unsigned)(row * HD);
#pragma unroll
  for (int hh = 0; hh < NH; ++hh)   // transposed store, fire-and-forget
    HBT[(size_t)(unsigned)((((b * NH + hh) * DM) + t) * NB + j)] =
        f2bf(hrow[hh * DM + t]);
  for (int hh = 0; hh < NH; ++hh) {
    red[t] = hrow[hh * DM + t] * att_src[hh * DM + t];
    __syncthreads();
    for (int off = 96; off >= 3; off >>= 1) {
      if (t < off) red[t] += red[t + off];
      __syncthreads();
    }
    if (t == 0) ASRC[(unsigned)(row * NH + hh)] = red[0] + red[1] + red[2];
    __syncthreads();
    red[t] = hrow[hh * DM + t] * att_dst[hh * DM + t];
    __syncthreads();
    for (int off = 96; off >= 3; off >>= 1) {
      if (t < off) red[t] += red[t + off];
      __syncthreads();
    }
    if (t == 0) ADST[(unsigned)(row * NH + hh)] = red[0] + red[1] + red[2];
    __syncthreads();
  }
}

// ---------------- softmax stats (online max/sum) ----------------
__global__ __launch_bounds__(128) void stats_kernel(
    const float* __restrict__ ASRC, const float* __restrict__ ADST,
    const unsigned* __restrict__ MASK, float* __restrict__ STATS) {
  int bi = blockIdx.x;          // b*NB + i
  int b = bi >> 9;
  int lane = threadIdx.x & 31, h = threadIdx.x >> 5;
  float adst = ADST[(unsigned)(bi * NH + h)];
  float m = -__builtin_inff(), s = 0.f;
  for (int j = lane; j < NB; j += 32) {
    unsigned w = MASK[(unsigned)(bi * 16 + (j >> 5))];
    if ((w >> (j & 31)) & 1u) {
      float sc = ASRC[(unsigned)((b * NB + j) * NH + h)] + adst;
      sc = sc >= 0.f ? sc : 0.2f * sc;
      float nm = fmaxf(m, sc);
      s = s * __expf(m - nm) + __expf(sc - nm);
      m = nm;
    }
  }
  for (int off = 16; off > 0; off >>= 1) {
    float om = __shfl_xor(m, off, 32);
    float os = __shfl_xor(s, off, 32);
    float nm = fmaxf(m, om);
    if (nm == -__builtin_inff()) { s = s + os; }
    else { s = s * __expf(m - nm) + os * __expf(om - nm); }
    m = nm;
  }
  if (lane == 0) {
    STATS[(unsigned)((bi * NH + h) * 2 + 0)] = m;
    STATS[(unsigned)((bi * NH + h) * 2 + 1)] = s;
  }
}

// ---------------- fused attention:  out = softmax(P) @ V  via WMMA ----------
// grid: (NB/16, NH, BATCH); block 128 = 4 waves; wave w covers cols w*48..+47.
// V (B operand) is loaded directly from the K-contiguous global HBT layout:
// per lane two aligned 16B loads give the 16 K-values of one column.
__global__ __launch_bounds__(128) void attn_kernel(
    const __bf16* __restrict__ HBT, const float* __restrict__ ASRC,
    const float* __restrict__ ADST, const float* __restrict__ STATS,
    const unsigned* __restrict__ MASK, float* __restrict__ HOUT) {
  int i0 = blockIdx.x * 16;
  int h  = blockIdx.y;
  int b  = blockIdx.z;
  int tid = threadIdx.x, lane = tid & 31, wave = tid >> 5;
  int half = lane >> 4, rl = lane & 15;
  __shared__ __bf16 Ps[16][APITCH];
  v8f zero = {0.f, 0.f, 0.f, 0.f, 0.f, 0.f, 0.f, 0.f};
  v8f acc0 = zero, acc1 = zero, acc2 = zero;

  int bBase = b * NB;
  const __bf16* vbase = HBT + (size_t)(unsigned)(((b * NH + h) * DM) * NB);
  // rows this thread fills in the P tile are fixed: rbase + 4q, column jj.
  int jj = tid & 31, rbase = tid >> 5;
  float negm[4], invs[4], adst[4];
#pragma unroll
  for (int q = 0; q < 4; ++q) {
    int i = i0 + rbase + 4 * q;
    negm[q] = -STATS[(unsigned)(((bBase + i) * NH + h) * 2 + 0)];
    invs[q] = 1.f / STATS[(unsigned)(((bBase + i) * NH + h) * 2 + 1)];
    adst[q] = ADST[(unsigned)((bBase + i) * NH + h)];
  }

  for (int j0 = 0; j0 < NB; j0 += 32) {
    float asrc_j = ASRC[(unsigned)((bBase + j0 + jj) * NH + h)];
#pragma unroll
    for (int q = 0; q < 4; ++q) {
      int i = i0 + rbase + 4 * q;
      unsigned w = MASK[(unsigned)((bBase + i) * 16 + (j0 >> 5))];
      float p = 0.f;
      if ((w >> jj) & 1u) {
        float sc = asrc_j + adst[q];
        sc = sc >= 0.f ? sc : 0.2f * sc;
        p = __expf(sc + negm[q]) * invs[q];
      }
      Ps[rbase + 4 * q][jj] = f2bf(p);
    }
    __syncthreads();
    v16bf a = mk16(*(const v8bf*)&Ps[rl][half * 16],
                   *(const v8bf*)&Ps[rl][half * 16 + 8]);
#pragma unroll
    for (int t = 0; t < 3; ++t) {
      int c = wave * 48 + t * 16 + rl;          // this lane's V column
      const __bf16* vp = vbase + (unsigned)(c * NB + j0 + half * 16);
      v16bf bv = mk16(*(const v8bf*)vp, *(const v8bf*)(vp + 8));
      v8f& acc = (t == 0) ? acc0 : (t == 1) ? acc1 : acc2;
      acc = __builtin_amdgcn_wmma_f32_16x16x32_bf16(false, a, false, bv,
                                                    (short)0, acc, false, false);
    }
    __syncthreads();
  }
#pragma unroll
  for (int t = 0; t < 3; ++t) {
    int c0 = wave * 48 + t * 16;
    v8f acc = (t == 0) ? acc0 : (t == 1) ? acc1 : acc2;
#pragma unroll
    for (int v = 0; v < 8; ++v) {
      int row = i0 + v + 8 * half;
      HOUT[(size_t)(unsigned)((bBase + row) * HD + h * DM + c0 + rl)] = acc[v];
    }
  }
}

// ---------------- head mean + bias + residual, also produces relu copy ------
__global__ __launch_bounds__(192) void gat_reduce(const float* __restrict__ HOUT,
                                                  const float* __restrict__ gatb,
                                                  float* __restrict__ X,
                                                  float* __restrict__ XR) {
  int row = blockIdx.x, t = threadIdx.x;
  const float* hr = HOUT + (size_t)(unsigned)(row * HD);
  float v = 0.25f * (hr[t] + hr[DM + t] + hr[2 * DM + t] + hr[3 * DM + t]) +
            gatb[t] + X[(size_t)(unsigned)(row * DM + t)];
  X[(size_t)(unsigned)(row * DM + t)] = v;
  XR[(size_t)(unsigned)(row * DM + t)] = fmaxf(v, 0.f);
}

__global__ __launch_bounds__(192) void gather_cls(const float* __restrict__ X,
                                                  float* __restrict__ out) {
  out[(unsigned)(blockIdx.x * DM + threadIdx.x)] =
      X[(size_t)(unsigned)((blockIdx.x * NB) * DM + threadIdx.x)];
}

__global__ __launch_bounds__(192) void pred_kernel(const float* __restrict__ A2,
                                                   const float* __restrict__ pw,
                                                   const float* __restrict__ pb,
                                                   float* __restrict__ out) {
  int b = blockIdx.x, t = threadIdx.x;
  __shared__ float rs[DM];
  rs[t] = A2[(unsigned)(b * DM + t)] * pw[t];
  __syncthreads();
  for (int off = 96; off >= 3; off >>= 1) {
    if (t < off) rs[t] += rs[t + off];
    __syncthreads();
  }
  if (t == 0) out[b] = rs[0] + rs[1] + rs[2] + pb[0];
}

// ---------------- host orchestration ----------------
static void launch_gemm(const float* A, int lda, const float* Bm, int ldb,
                        float* C, int ldc, int M, int N, int K,
                        const float* bias, const float* r1, const float* r2,
                        int relu, hipStream_t s) {
  dim3 g(N / TN, M / TM);
  if (bias && r1 && r2)
    gemm_bf16_wmma<1, 1, 1, 0><<<g, 256, 0, s>>>(A, lda, Bm, ldb, C, ldc, K, bias, r1, r2);
  else if (bias && relu)
    gemm_bf16_wmma<1, 0, 0, 1><<<g, 256, 0, s>>>(A, lda, Bm, ldb, C, ldc, K, bias, r1, r2);
  else
    gemm_bf16_wmma<0, 0, 0, 0><<<g, 256, 0, s>>>(A, lda, Bm, ldb, C, ldc, K, bias, r1, r2);
}

extern "C" void kernel_launch(void* const* d_in, const int* in_sizes, int n_in,
                              void* d_out, int out_size, void* d_ws, size_t ws_size,
                              hipStream_t stream) {
  // flat input order (insertion-order recursive over setup_inputs dict)
  const float* x      = (const float*)d_in[0];
  const float* adj    = (const float*)d_in[1];
  const int*   depth  = (const int*)d_in[2];
  const float* clsp   = (const float*)d_in[3];
  const float* opW    = (const float*)d_in[4];
  const float* opb    = (const float*)d_in[5];
  const float* depW   = (const float*)d_in[6];
  const float* depb   = (const float*)d_in[7];
  const float* prevg  = (const float*)d_in[8];
  const float* prevb  = (const float*)d_in[9];
  struct Layer {
    const float *ng, *nb, *gW, *asrc, *adst, *gb, *fng, *fnb, *W1, *b1, *W2, *b2;
  } L[2];
  for (int l = 0; l < 2; ++l) {
    int o = 10 + l * 12;
    L[l].ng = (const float*)d_in[o + 0];  L[l].nb = (const float*)d_in[o + 1];
    L[l].gW = (const float*)d_in[o + 2];  L[l].asrc = (const float*)d_in[o + 3];
    L[l].adst = (const float*)d_in[o + 4]; L[l].gb = (const float*)d_in[o + 5];
    L[l].fng = (const float*)d_in[o + 6]; L[l].fnb = (const float*)d_in[o + 7];
    L[l].W1 = (const float*)d_in[o + 8];  L[l].b1 = (const float*)d_in[o + 9];
    L[l].W2 = (const float*)d_in[o + 10]; L[l].b2 = (const float*)d_in[o + 11];
  }
  const float* postg = (const float*)d_in[34];
  const float* postb = (const float*)d_in[35];
  const float* h1W   = (const float*)d_in[36];
  const float* h1b   = (const float*)d_in[37];
  const float* h2W   = (const float*)d_in[38];
  const float* h2b   = (const float*)d_in[39];
  const float* predW = (const float*)d_in[40];
  const float* predb = (const float*)d_in[41];
  float* out = (float*)d_out;

  // workspace carve
  char* ws = (char*)d_ws;
  auto carve = [&](size_t bytes) {
    void* p = (void*)ws;
    ws += (bytes + 255) & ~(size_t)255;
    return p;
  };
  float*    X     = (float*)carve((size_t)ROWS * DM * 4);
  float*    T1    = (float*)carve((size_t)ROWS * DM * 4);
  float*    T2    = (float*)carve((size_t)ROWS * DM * 4);
  float*    Hbuf  = (float*)carve((size_t)ROWS * HD * 4);   // h / attn-out / ffn mid
  __bf16*   HBT   = (__bf16*)carve((size_t)ROWS * HD * 2);  // V, K-contiguous
  float*    ASRC  = (float*)carve((size_t)ROWS * NH * 4);
  float*    ADST  = (float*)carve((size_t)ROWS * NH * 4);
  float*    STATS = (float*)carve((size_t)ROWS * NH * 2 * 4);
  unsigned* MASK  = (unsigned*)carve((size_t)ROWS * 16 * 4);
  float*    CRAW  = (float*)carve((size_t)BATCH * DM * 4);
  float*    CLN   = (float*)carve((size_t)BATCH * DM * 4);
  float*    A1    = (float*)carve((size_t)BATCH * DM * 4);
  float*    A2    = (float*)carve((size_t)BATCH * DM * 4);

  mask_kernel<<<ROWS, 32, 0, stream>>>(adj, MASK);
  embed_kernel<<<BATCH * NRAW, DM, 0, stream>>>(x, depth, opW, opb, depW, depb,
                                                clsp, X);
  ln_kernel<<<ROWS, DM, 0, stream>>>(X, X, prevg, prevb);   // prev LN

  for (int l = 0; l < 2; ++l) {
    // --- GAT block ---
    ln_kernel<<<ROWS, DM, 0, stream>>>(X, T1, L[l].ng, L[l].nb);
    launch_gemm(T1, DM, L[l].gW, HD, Hbuf, HD, ROWS, HD, DM,
                nullptr, nullptr, nullptr, 0, stream);          // h = x_ln @ W
    prep_attn<<<ROWS, DM, 0, stream>>>(Hbuf, L[l].asrc, L[l].adst, HBT, ASRC, ADST);
    stats_kernel<<<ROWS, 128, 0, stream>>>(ASRC, ADST, MASK, STATS);
    attn_kernel<<<dim3(NB / 16, NH, BATCH), 128, 0, stream>>>(HBT, ASRC, ADST,
                                                              STATS, MASK, Hbuf);
    gat_reduce<<<ROWS, DM, 0, stream>>>(Hbuf, L[l].gb, X, T1);  // X+=gat, T1=relu(X)
    // --- FFN block ---
    ln_kernel<<<ROWS, DM, 0, stream>>>(T1, T2, L[l].fng, L[l].fnb);
    launch_gemm(T2, DM, L[l].W1, HD, Hbuf, HD, ROWS, HD, DM,
                L[l].b1, nullptr, nullptr, 1, stream);          // relu(t@W1+b1)
    launch_gemm(Hbuf, HD, L[l].W2, DM, X, DM, ROWS, DM, HD,
                L[l].b2, T1, X, 0, stream);                     // X = .. + xr + X
  }

  gather_cls<<<BATCH, DM, 0, stream>>>(X, CRAW);
  ln_kernel<<<BATCH, DM, 0, stream>>>(CRAW, CLN, postg, postb);
  launch_gemm(CLN, DM, h1W, DM, A1, DM, BATCH, DM, DM, h1b, nullptr, nullptr, 1, stream);
  launch_gemm(A1, DM, h2W, DM, A2, DM, BATCH, DM, DM, h2b, nullptr, nullptr, 1, stream);
  pred_kernel<<<BATCH, DM, 0, stream>>>(A2, predW, predb, out);
  (void)in_sizes; (void)n_in; (void)out_size; (void)ws_size;
}
